// LearnableWaveletEnhanced_80985903333552
// MI455X (gfx1250) — matrix operand order repaired and checked by reading, not compiled, not run on hardware
//
#include <hip/hip_runtime.h>
#include <hip/hip_bf16.h>

typedef __attribute__((ext_vector_type(16))) _Float16 v16h;
typedef __attribute__((ext_vector_type(8)))  _Float16 v8h;
typedef __attribute__((ext_vector_type(8)))  float    v8f;

#define B_   16
#define C_   32
#define L_   1024
#define D_   256
#define H_   4
#define HD_  64

// ---------------------------------------------------------------------------
// WMMA fragment loaders (wave32, 16x16x32 f16).
// A-matrix 16x32 f16: lane = row(0..15) + 16*hf ; VGPR v holds K pair
//   k = (v/4)*16 + hf*8 + (v%4)*2  -> two contiguous 16B runs per lane.
// B-matrix 32x16 f16: lane = col(0..15) + 16*hf ; same K order as A:
//   k = (v/4)*16 + hf*8 + (v%4)*2 -> two contiguous 16B runs per lane when
//   the source is stored (N,K) row-major (i.e. torch-style weight layout).
// ---------------------------------------------------------------------------
__device__ __forceinline__ v16h load_a_frag(const _Float16* A, int lda) {
  int lane = threadIdx.x & 31;
  int row = lane & 15, hf = lane >> 4;
  const _Float16* p = A + (size_t)row * lda + hf * 8;
  v8h lo = *(const v8h*)p;
  v8h hi = *(const v8h*)(p + 16);
  v16h a;
#pragma unroll
  for (int i = 0; i < 8; ++i) { a[i] = lo[i]; a[8 + i] = hi[i]; }
  return a;
}

__device__ __forceinline__ v16h load_b_frag(const _Float16* Bt, int ldb) {
  int lane = threadIdx.x & 31;
  int col = lane & 15, hf = lane >> 4;
  const _Float16* p = Bt + (size_t)col * ldb + hf * 8;
  v8h lo = *(const v8h*)p;
  v8h hi = *(const v8h*)(p + 16);
  v16h b;
#pragma unroll
  for (int i = 0; i < 8; ++i) { b[i] = lo[i]; b[8 + i] = hi[i]; }
  return b;
}

__device__ __forceinline__ v8f wmma_f16(v16h a, v16h b, v8f c) {
  return __builtin_amdgcn_wmma_f32_16x16x32_f16(false, a, false, b, (short)0, c,
                                                false, false);
}

// ---------------------------------------------------------------------------
// Wavelet level: one stride-2 conv with the softmax-weighted effective filter.
// grid = B*C blocks over rows of (B*C, Lc).
// ---------------------------------------------------------------------------
__global__ void wavelet_level_kernel(const float* __restrict__ in,
                                     float* __restrict__ lo_out,
                                     float* __restrict__ hi_out,
                                     const float* __restrict__ lofilt,
                                     const float* __restrict__ hifilt,
                                     const float* __restrict__ fw, int Lc) {
  // softmax over 8 filter weights -> effective 8-tap filters
  float w[8];
  float mx = -1e30f;
#pragma unroll
  for (int f = 0; f < 8; ++f) mx = fmaxf(mx, fw[f]);
  float s = 0.f;
#pragma unroll
  for (int f = 0; f < 8; ++f) { w[f] = __expf(fw[f] - mx); s += w[f]; }
  float inv = 1.0f / s;
  float fl[8], fh[8];
#pragma unroll
  for (int k = 0; k < 8; ++k) {
    float al = 0.f, ah = 0.f;
#pragma unroll
    for (int f = 0; f < 8; ++f) {
      al += w[f] * lofilt[f * 8 + k];
      ah += w[f] * hifilt[f * 8 + k];
    }
    fl[k] = al * inv; fh[k] = ah * inv;
  }
  int rc = blockIdx.x;
  const float* row = in + (size_t)rc * Lc;
  int Lo = Lc >> 1;
  for (int j = threadIdx.x; j < Lo; j += blockDim.x) {
    float al = 0.f, ah = 0.f;
    int base = 2 * j - 3;
#pragma unroll
    for (int k = 0; k < 8; ++k) {
      int idx = base + k;
      float xv = (idx >= 0 && idx < Lc) ? row[idx] : 0.f;
      al += xv * fl[k];
      ah += xv * fh[k];
    }
    lo_out[(size_t)rc * Lo + j] = al;
    hi_out[(size_t)rc * Lo + j] = ah;
  }
}

// ---------------------------------------------------------------------------
// im2col for the K=3 gate conv: (B*L, C*3) in f16 so the conv becomes a GEMM.
// ---------------------------------------------------------------------------
__global__ void im2col_kernel(const float* __restrict__ x,
                              _Float16* __restrict__ xcol) {
  int idx = blockIdx.x * blockDim.x + threadIdx.x;
  if (idx >= B_ * L_ * 96) return;
  int m = idx / 96, r = idx % 96;
  int c = r / 3, k = r % 3;
  int b = m >> 10, l = m & 1023;
  int li = l + k - 1;
  float v = (li >= 0 && li < L_) ? x[((size_t)b * C_ + c) * L_ + li] : 0.f;
  xcol[idx] = (_Float16)v;
}

__global__ void convert_f16_kernel(const float* __restrict__ in,
                                   _Float16* __restrict__ out, int n) {
  int i = blockIdx.x * blockDim.x + threadIdx.x;
  if (i < n) out[i] = (_Float16)in[i];
}

__global__ void zero_kernel(float* __restrict__ p, int n) {
  int i = blockIdx.x * blockDim.x + threadIdx.x;
  if (i < n) p[i] = 0.f;
}

// ---------------------------------------------------------------------------
// Generic WMMA GEMM: C[M,N] = A[M,K](f16,row) * W[N,K](f16,row)^T + bias.
// block = 128 (4 waves, one 16x16 tile each), grid = (N/16, M/64).
// mode 0: store f16 row-major (M,N)   [gate-conv feat]
// mode 1: QKV scatter: q,k as (b,h,l,hd); v transposed as (b,h,hd,l)
// ---------------------------------------------------------------------------
__global__ void wmma_gemm_kernel(const _Float16* __restrict__ A,
                                 const _Float16* __restrict__ W,
                                 const float* __restrict__ bias, int M, int N,
                                 int K, _Float16* __restrict__ Cout,
                                 _Float16* __restrict__ qo,
                                 _Float16* __restrict__ ko,
                                 _Float16* __restrict__ vo, int mode) {
  int wave = threadIdx.x >> 5;
  int m0 = blockIdx.y * 64 + wave * 16;
  int n0 = blockIdx.x * 16;
  if (m0 >= M) return;
  v8f acc;
#pragma unroll
  for (int i = 0; i < 8; ++i) acc[i] = 0.f;
  const _Float16* Ab = A + (size_t)m0 * K;
  const _Float16* Wb = W + (size_t)n0 * K;
  for (int kb = 0; kb < K; kb += 32) {
    if (kb + 32 < K) {
      __builtin_prefetch(Ab + kb + 32, 0, 1);
      __builtin_prefetch(Wb + kb + 32, 0, 1);
    }
    v16h af = load_a_frag(Ab + kb, K);
    v16h bf = load_b_frag(Wb + kb, K);
    acc = wmma_f16(af, bf, acc);
  }
  int lane = threadIdx.x & 31;
  int col = lane & 15, hf = lane >> 4;
#pragma unroll
  for (int v = 0; v < 8; ++v) {
    int row = m0 + v + 8 * hf;
    int n = n0 + col;
    float val = acc[v] + bias[n];
    _Float16 hv = (_Float16)val;
    if (mode == 0) {
      Cout[(size_t)row * N + n] = hv;
    } else {
      int b = row >> 10, l = row & 1023;
      if (n < 256) {
        int h = n >> 6, d = n & 63;
        qo[(((size_t)(b * H_ + h)) * L_ + l) * HD_ + d] = hv;
      } else if (n < 512) {
        int nn = n - 256, h = nn >> 6, d = nn & 63;
        ko[(((size_t)(b * H_ + h)) * L_ + l) * HD_ + d] = hv;
      } else {
        int nn = n - 512, h = nn >> 6, d = nn & 63;
        vo[(((size_t)(b * H_ + h)) * HD_ + d) * L_ + l] = hv;  // transposed
      }
    }
  }
}

// ---------------------------------------------------------------------------
// Flash attention, one wave per (b,h, 16-row Q tile). Online softmax.
// S = QK^T via 16x16x32 WMMA (hd=64 -> 2 steps); P staged through LDS to
// build the A-fragment; O += P*V via WMMA against V^T. O rows are mean-
// reduced into obar[b*256 + h*64 + d] with f32 atomics (mean commutes with
// out_proj, so full O never hits memory).
// ---------------------------------------------------------------------------
__global__ void flash_attn_kernel(const _Float16* __restrict__ Q,
                                  const _Float16* __restrict__ Kt,
                                  const _Float16* __restrict__ Vt,
                                  float* __restrict__ obar) {
  int bh = blockIdx.y;
  int q0 = blockIdx.x * 16;
  int lane = threadIdx.x & 31;
  int col = lane & 15, hf = lane >> 4;
  __shared__ __align__(16) _Float16 Pl[16 * 32];

  const _Float16* Qb = Q + ((size_t)bh * L_ + q0) * HD_;
  v16h aq0 = load_a_frag(Qb, HD_);
  v16h aq1 = load_a_frag(Qb + 32, HD_);

  v8f acc0, acc1, acc2, acc3;
  float mrow[8], lrow[8];
#pragma unroll
  for (int v = 0; v < 8; ++v) {
    acc0[v] = 0.f; acc1[v] = 0.f; acc2[v] = 0.f; acc3[v] = 0.f;
    mrow[v] = -1e30f; lrow[v] = 0.f;
  }
  const float sc = 0.125f;  // 1/sqrt(64)

  for (int kb = 0; kb < L_; kb += 32) {
    const _Float16* Kb0 = Kt + ((size_t)bh * L_ + kb) * HD_;
    const _Float16* Kb1 = Kb0 + 16 * HD_;
    v8f s0, s1;
#pragma unroll
    for (int i = 0; i < 8; ++i) { s0[i] = 0.f; s1[i] = 0.f; }
    s0 = wmma_f16(aq0, load_b_frag(Kb0, HD_), s0);
    s0 = wmma_f16(aq1, load_b_frag(Kb0 + 32, HD_), s0);
    s1 = wmma_f16(aq0, load_b_frag(Kb1, HD_), s1);
    s1 = wmma_f16(aq1, load_b_frag(Kb1 + 32, HD_), s1);

#pragma unroll
    for (int v = 0; v < 8; ++v) {
      float a = s0[v] * sc, b = s1[v] * sc;
      float mx = fmaxf(a, b);
#pragma unroll
      for (int off = 1; off < 16; off <<= 1)
        mx = fmaxf(mx, __shfl_xor(mx, off, 32));
      float mnew = fmaxf(mrow[v], mx);
      float rescale = __expf(mrow[v] - mnew);
      mrow[v] = mnew;
      float p0 = __expf(a - mnew), p1 = __expf(b - mnew);
      float rs = p0 + p1;
#pragma unroll
      for (int off = 1; off < 16; off <<= 1) rs += __shfl_xor(rs, off, 32);
      lrow[v] = lrow[v] * rescale + rs;
      acc0[v] *= rescale; acc1[v] *= rescale;
      acc2[v] *= rescale; acc3[v] *= rescale;
      int r = v + 8 * hf;
      Pl[r * 32 + col] = (_Float16)p0;
      Pl[r * 32 + 16 + col] = (_Float16)p1;
    }
    __syncthreads();
    {
      int row = lane & 15;
      const _Float16* pp = &Pl[row * 32 + hf * 8];
      v8h plo = *(const v8h*)pp;
      v8h phi = *(const v8h*)(pp + 16);
      v16h ap;
#pragma unroll
      for (int i = 0; i < 8; ++i) { ap[i] = plo[i]; ap[8 + i] = phi[i]; }
      __syncthreads();
      const _Float16* Vb = Vt + (size_t)bh * HD_ * L_ + kb;  // (d, L) layout
      acc0 = wmma_f16(ap, load_b_frag(Vb + (size_t)0 * 16 * L_, L_), acc0);
      acc1 = wmma_f16(ap, load_b_frag(Vb + (size_t)1 * 16 * L_, L_), acc1);
      acc2 = wmma_f16(ap, load_b_frag(Vb + (size_t)2 * 16 * L_, L_), acc2);
      acc3 = wmma_f16(ap, load_b_frag(Vb + (size_t)3 * 16 * L_, L_), acc3);
    }
  }

  int b = bh >> 2, h = bh & 3;
  float* ob = obar + b * D_ + h * HD_;
#pragma unroll
  for (int v = 0; v < 8; ++v) {
    float il = 1.0f / lrow[v];
    atomicAdd(&ob[0 * 16 + col], acc0[v] * il);
    atomicAdd(&ob[1 * 16 + col], acc1[v] * il);
    atomicAdd(&ob[2 * 16 + col], acc2[v] * il);
    atomicAdd(&ob[3 * 16 + col], acc3[v] * il);
  }
}

// ---------------------------------------------------------------------------
// Tiny gate MLP head: g = mean(O); out_proj -> fc1(relu) -> fc2(sigmoid).
// One block of 256 threads, loops over the 16 batches.
// ---------------------------------------------------------------------------
__global__ void gate_mlp_kernel(const float* __restrict__ obar,
                                const float* __restrict__ Wo,
                                const float* __restrict__ bo,
                                const float* __restrict__ W1,
                                const float* __restrict__ b1,
                                const float* __restrict__ W2,
                                const float* __restrict__ b2,
                                float* __restrict__ gates) {
  __shared__ float sm[D_], sg[D_];
  int t = threadIdx.x;
  for (int b = 0; b < B_; ++b) {
    sm[t] = obar[b * D_ + t] * (1.0f / (float)L_);
    __syncthreads();
    float a = bo[t];
    for (int d = 0; d < D_; ++d) a += Wo[t * D_ + d] * sm[d];
    sg[t] = a;
    __syncthreads();
    float a1 = b1[t];
    for (int d = 0; d < D_; ++d) a1 += W1[t * D_ + d] * sg[d];
    a1 = fmaxf(a1, 0.f);
    __syncthreads();
    sm[t] = a1;
    __syncthreads();
    if (t < 6) {
      float a2 = b2[t];
      for (int d = 0; d < D_; ++d) a2 += W2[t * D_ + d] * sm[d];
      gates[b * 6 + t] = 1.0f / (1.0f + __expf(-a2));
    }
    __syncthreads();
  }
}

// ---------------------------------------------------------------------------
// Apply gates and pack outputs: approx(16,32,128) | d0(16,32,512) |
// d1(16,32,256) | d2(16,32,128), flat in return order.
// ---------------------------------------------------------------------------
__global__ void output_kernel(const float* __restrict__ lo3,
                              const float* __restrict__ hi1,
                              const float* __restrict__ hi2,
                              const float* __restrict__ hi3,
                              const float* __restrict__ gates,
                              float* __restrict__ out) {
  int i = blockIdx.x * blockDim.x + threadIdx.x;
  const int nA = B_ * C_ * 128;   // 65536
  const int n0 = B_ * C_ * 512;   // 262144
  const int n1 = B_ * C_ * 256;   // 131072
  const int n2 = B_ * C_ * 128;   // 65536
  if (i < nA) {
    int b = i / (C_ * 128);
    out[i] = lo3[i] * gates[b * 6 + 4];            // gates[:, :, 2, 0]
  } else if (i < nA + n0) {
    int j = i - nA, b = j / (C_ * 512);
    out[i] = hi1[j] * gates[b * 6 + 1];            // gates[:, :, 0, 1]
  } else if (i < nA + n0 + n1) {
    int j = i - nA - n0, b = j / (C_ * 256);
    out[i] = hi2[j] * gates[b * 6 + 3];            // gates[:, :, 1, 1]
  } else if (i < nA + n0 + n1 + n2) {
    int j = i - nA - n0 - n1, b = j / (C_ * 128);
    out[i] = hi3[j] * gates[b * 6 + 5];            // gates[:, :, 2, 1]
  }
}

// ---------------------------------------------------------------------------
extern "C" void kernel_launch(void* const* d_in, const int* in_sizes, int n_in,
                              void* d_out, int out_size, void* d_ws,
                              size_t ws_size, hipStream_t stream) {
  const float* x      = (const float*)d_in[0];
  const float* lof    = (const float*)d_in[1];
  const float* hif    = (const float*)d_in[2];
  const float* fw     = (const float*)d_in[3];
  const float* gcw    = (const float*)d_in[4];
  const float* gcb    = (const float*)d_in[5];
  const float* ipw    = (const float*)d_in[6];
  const float* ipb    = (const float*)d_in[7];
  const float* opw    = (const float*)d_in[8];
  const float* opb    = (const float*)d_in[9];
  const float* f1w    = (const float*)d_in[10];
  const float* f1b    = (const float*)d_in[11];
  const float* f2w    = (const float*)d_in[12];
  const float* f2b    = (const float*)d_in[13];
  float* out = (float*)d_out;

  char* w = (char*)d_ws;
  size_t off = 0;
  auto alloc = [&](size_t bytes) {
    size_t o = off;
    off = (off + bytes + 255) & ~(size_t)255;
    return o;
  };
  float*    hi1   = (float*)(w + alloc((size_t)B_ * C_ * 512 * 4));
  float*    hi2   = (float*)(w + alloc((size_t)B_ * C_ * 256 * 4));
  float*    hi3   = (float*)(w + alloc((size_t)B_ * C_ * 128 * 4));
  float*    lo1   = (float*)(w + alloc((size_t)B_ * C_ * 512 * 4));
  float*    lo2   = (float*)(w + alloc((size_t)B_ * C_ * 256 * 4));
  float*    lo3   = (float*)(w + alloc((size_t)B_ * C_ * 128 * 4));
  _Float16* xcol  = (_Float16*)(w + alloc((size_t)B_ * L_ * 96 * 2));
  _Float16* wg16  = (_Float16*)(w + alloc((size_t)D_ * 96 * 2));
  _Float16* wq16  = (_Float16*)(w + alloc((size_t)3 * D_ * D_ * 2));
  _Float16* feat  = (_Float16*)(w + alloc((size_t)B_ * L_ * D_ * 2));
  _Float16* q16   = (_Float16*)(w + alloc((size_t)B_ * H_ * L_ * HD_ * 2));
  _Float16* k16   = (_Float16*)(w + alloc((size_t)B_ * H_ * L_ * HD_ * 2));
  _Float16* vt16  = (_Float16*)(w + alloc((size_t)B_ * H_ * L_ * HD_ * 2));
  float*    obar  = (float*)(w + alloc((size_t)B_ * D_ * 4));
  float*    gates = (float*)(w + alloc((size_t)B_ * 6 * 4));

  // 1) wavelet pyramid (effective-filter trick)
  wavelet_level_kernel<<<B_ * C_, 256, 0, stream>>>(x, lo1, hi1, lof, hif, fw, 1024);
  wavelet_level_kernel<<<B_ * C_, 256, 0, stream>>>(lo1, lo2, hi2, lof, hif, fw, 512);
  wavelet_level_kernel<<<B_ * C_, 256, 0, stream>>>(lo2, lo3, hi3, lof, hif, fw, 256);

  // 2) im2col + weight conversion for the WMMA GEMMs
  {
    int n = B_ * L_ * 96;
    im2col_kernel<<<(n + 255) / 256, 256, 0, stream>>>(x, xcol);
    int n1 = D_ * 96;
    convert_f16_kernel<<<(n1 + 255) / 256, 256, 0, stream>>>(gcw, wg16, n1);
    int n2 = 3 * D_ * D_;
    convert_f16_kernel<<<(n2 + 255) / 256, 256, 0, stream>>>(ipw, wq16, n2);
  }

  // 3) gate conv as WMMA GEMM: feat(16384,256) = xcol(16384,96) * Wg^T
  {
    dim3 grid(D_ / 16, (B_ * L_) / 64);
    wmma_gemm_kernel<<<grid, 128, 0, stream>>>(xcol, wg16, gcb, B_ * L_, D_, 96,
                                               feat, nullptr, nullptr, nullptr, 0);
  }
  // 4) QKV projection as WMMA GEMM with scatter epilogue
  {
    dim3 grid((3 * D_) / 16, (B_ * L_) / 64);
    wmma_gemm_kernel<<<grid, 128, 0, stream>>>(feat, wq16, ipb, B_ * L_, 3 * D_,
                                               D_, nullptr, q16, k16, vt16, 1);
  }

  // 5) flash attention, accumulating mean(O) directly
  zero_kernel<<<(B_ * D_ + 255) / 256, 256, 0, stream>>>(obar, B_ * D_);
  {
    dim3 grid(L_ / 16, B_ * H_);
    flash_attn_kernel<<<grid, 32, 0, stream>>>(q16, k16, vt16, obar);
  }

  // 6) gate MLP head -> sigmoid gates (B,6)
  gate_mlp_kernel<<<1, 256, 0, stream>>>(obar, opw, opb, f1w, f1b, f2w, f2b, gates);

  // 7) gate the wavelet coefficients into the packed output
  {
    int n = out_size;
    output_kernel<<<(n + 255) / 256, 256, 0, stream>>>(lo3, hi1, hi2, hi3, gates, out);
  }
}
